// TableCL_80796924772452
// MI455X (gfx1250) — compile-verified, matrix operand order checked by reading, not run on hardware
//
#include <hip/hip_runtime.h>
#include <hip/hip_bf16.h>
#include <math.h>

// ---------------------------------------------------------------------------
// SupCon loss for MI455X (gfx1250, wave32, WMMA).
// Flash-style single pass over the Gram matrix feat@feat^T using
// V_WMMA_F32_16X16X32_BF16 with an fp32 -> (hi,lo) bf16 split
// (3 WMMAs per K=32 step ~ full fp32 accuracy on the logits).
// ---------------------------------------------------------------------------

typedef __attribute__((ext_vector_type(16))) __bf16 v16bf;
typedef __attribute__((ext_vector_type(8)))  float  v8f;

#define B_ 8
#define T_ 1024
#define D_ 1024
#define NEG_BIG  (-3.0e38f)
#define INV_TEMP 10.0f   // 1 / TEMPERATURE

union FragU { v16bf v; uint4 u[2]; };

// 16-bit A/B fragment layout (ISA 7.12.2): lane (l = lane&15) holds row/col
// (base + l); half h = lane>>4 selects K chunks [k0+8h, k0+8h+8) and
// [k0+16+8h, k0+16+8h+8)  -> two 16-byte loads per lane.
__device__ __forceinline__ v16bf load_frag(const __bf16* __restrict__ rowAtK0, int h) {
  FragU f;
  f.u[0] = *(const uint4*)(rowAtK0 + 8 * h);
  f.u[1] = *(const uint4*)(rowAtK0 + 16 + 8 * h);
  return f.v;
}

// ---------------- kernel 1: f32 -> hi/lo bf16 split ------------------------
__global__ void supcon_cvt_kernel(const float* __restrict__ feat,
                                  __bf16* __restrict__ hi,
                                  __bf16* __restrict__ lo,
                                  long n) {
  long i = (long)blockIdx.x * blockDim.x + threadIdx.x;
  if (i < n) {
    float x = feat[i];
    __bf16 h = (__bf16)x;            // RNE truncation to bf16
    float  r = x - (float)h;         // residual: next ~8 mantissa bits
    hi[i] = h;
    lo[i] = (__bf16)r;
  }
}

// ---------------- kernel 2: zero accumulators ------------------------------
__global__ void supcon_init_kernel(float* __restrict__ lossAcc, int* __restrict__ valid) {
  int i = threadIdx.x;
  if (i < B_) { lossAcc[i] = 0.0f; valid[i] = 0; }
}

// ---------------- kernel 3: per-batch mask sanity check --------------------
// valid_b = any over (i<L, j<L) of mask[i,j] != ((i==j && i!=0) ? 1 : 0)
__global__ void supcon_valid_kernel(const float* __restrict__ masks,
                                    const int* __restrict__ icl,
                                    int* __restrict__ valid) {
  int b = blockIdx.y;
  int L = icl[b] + 1;
  const float* mb = masks + (size_t)b * T_ * T_;
  long tid = (long)blockIdx.x * blockDim.x + threadIdx.x;
  long stride = (long)gridDim.x * blockDim.x;
  int found = 0;
  for (long e = tid; e < (long)T_ * T_; e += stride) {
    int i = (int)(e >> 10);
    int j = (int)(e & (T_ - 1));
    if (i < L && j < L) {
      float expect = (i == j && i != 0) ? 1.0f : 0.0f;
      if (mb[e] != expect) { found = 1; break; }
    }
  }
  if (found) atomicOr(&valid[b], 1);
}

// ---------------- kernel 4: main flash-supcon kernel -----------------------
// Grid (T/16, B); 256 threads = 8 waves. All waves share one 16-row block;
// each wave walks a strided set of 16-col tiles with a full K=1024 WMMA loop,
// maintaining online (max, S) plus linear (A, C) per row.
__global__ __launch_bounds__(256) void supcon_main_kernel(
    const __bf16* __restrict__ hiA, const __bf16* __restrict__ loA,
    const float* __restrict__ masks, const int* __restrict__ icl,
    float* __restrict__ lossAcc) {
  int b = blockIdx.y;
  int L = icl[b] + 1;
  int row0 = blockIdx.x * 16;
  if (row0 >= L) return;   // rows >= L contribute nothing

  const int tid  = threadIdx.x;
  const int wave = tid >> 5;
  const int lane = tid & 31;
  const int l    = lane & 15;
  const int h    = lane >> 4;

  const __bf16* hiBase = hiA + (size_t)b * T_ * D_;
  const __bf16* loBase = loA + (size_t)b * T_ * D_;
  const float*  maskB  = masks + (size_t)b * T_ * T_;

  // A-matrix rows for this block (lane l -> row row0+l)
  const __bf16* aHiRow = hiBase + (size_t)(row0 + l) * D_;
  const __bf16* aLoRow = loBase + (size_t)(row0 + l) * D_;

  // Per-row running stats. C-layout: VGPR v holds row (v + 8h) across the
  // 16-lane half-group, so stats are kept uniform within each half via
  // width-16 butterfly reductions.
  float runMax[8], runS[8], accA[8], accC[8];
#pragma unroll
  for (int v = 0; v < 8; ++v) { runMax[v] = NEG_BIG; runS[v] = 0.0f; accA[v] = 0.0f; accC[v] = 0.0f; }

  const int nct = (L + 15) >> 4;   // column tiles covering valid cols
  for (int ct = wave; ct < nct; ct += 8) {
    const int col0 = ct << 4;
    const __bf16* bHiRow = hiBase + (size_t)(col0 + l) * D_;
    const __bf16* bLoRow = loBase + (size_t)(col0 + l) * D_;

    v8f acc = {0.f, 0.f, 0.f, 0.f, 0.f, 0.f, 0.f, 0.f};
    for (int k0 = 0; k0 < D_; k0 += 32) {
      v16bf ah = load_frag(aHiRow + k0, h);
      v16bf al = load_frag(aLoRow + k0, h);
      v16bf bh = load_frag(bHiRow + k0, h);
      v16bf bl = load_frag(bLoRow + k0, h);
      // fp32-split product: hi*hi + hi*lo + lo*hi (lo*lo ~ 2^-16 rel, dropped)
      acc = __builtin_amdgcn_wmma_f32_16x16x32_bf16(false, ah, false, bh, (short)0, acc, false, false);
      acc = __builtin_amdgcn_wmma_f32_16x16x32_bf16(false, ah, false, bl, (short)0, acc, false, false);
      acc = __builtin_amdgcn_wmma_f32_16x16x32_bf16(false, al, false, bh, (short)0, acc, false, false);
    }

    const int  col = col0 + l;
    const bool cv  = (col < L);
#pragma unroll
    for (int v = 0; v < 8; ++v) {
      float c   = acc[v] * INV_TEMP;          // logits before max subtraction
      int   row = row0 + v + 8 * h;
      // row max over valid cols (diagonal included, per reference)
      float tm = cv ? c : NEG_BIG;
      tm = fmaxf(tm, __shfl_xor(tm, 8, 16));
      tm = fmaxf(tm, __shfl_xor(tm, 4, 16));
      tm = fmaxf(tm, __shfl_xor(tm, 2, 16));
      tm = fmaxf(tm, __shfl_xor(tm, 1, 16));
      float newmax = fmaxf(runMax[v], tm);
      // S excludes the diagonal (logits_mask)
      bool  inc  = cv && (col != row);
      float mval = inc ? maskB[(size_t)row * T_ + col] : 0.0f;
      float e    = inc ? __expf(c - newmax) : 0.0f;
      e += __shfl_xor(e, 8, 16);
      e += __shfl_xor(e, 4, 16);
      e += __shfl_xor(e, 2, 16);
      e += __shfl_xor(e, 1, 16);
      runS[v]   = runS[v] * __expf(runMax[v] - newmax) + e;
      runMax[v] = newmax;
      accA[v]  += mval * c;                   // linear in dot -> no rescale
      accC[v]  += mval;
    }
  }

  // reduce the linear accumulators across the 16-lane half-group
#pragma unroll
  for (int v = 0; v < 8; ++v) {
    for (int off = 8; off >= 1; off >>= 1) {
      accA[v] += __shfl_xor(accA[v], off, 16);
      accC[v] += __shfl_xor(accC[v], off, 16);
    }
  }

  // cross-wave combine through LDS
  __shared__ float sM[8][16], sS[8][16], sA[8][16], sC[8][16];
  if (l == 0) {
#pragma unroll
    for (int v = 0; v < 8; ++v) {
      int r = v + 8 * h;
      sM[wave][r] = runMax[v]; sS[wave][r] = runS[v];
      sA[wave][r] = accA[v];   sC[wave][r] = accC[v];
    }
  }
  __syncthreads();

  if (tid < 16) {
    int   r = tid;
    float M = NEG_BIG, S = 0.0f, A = 0.0f, C = 0.0f;
    for (int w = 0; w < 8; ++w) {
      float mw = sM[w][r];
      float nm = fmaxf(M, mw);
      S = S * __expf(M - nm) + sS[w][r] * __expf(mw - nm);
      M = nm;
      A += sA[w][r];
      C += sC[w][r];
    }
    int   row = row0 + r;
    float contrib = 0.0f;
    if (row < L) {
      // mean_log_prob_pos = (A - C*M - C*log(S+1e-6)) / (C + 1e-6)
      float mlpp = (A - C * M - C * __logf(S + 1e-6f)) / (C + 1e-6f);
      contrib = -mlpp / (float)L;
    }
    contrib += __shfl_xor(contrib, 8, 16);
    contrib += __shfl_xor(contrib, 4, 16);
    contrib += __shfl_xor(contrib, 2, 16);
    contrib += __shfl_xor(contrib, 1, 16);
    if (tid == 0) atomicAdd(&lossAcc[b], contrib);
  }
}

// ---------------- kernel 5: finalize ---------------------------------------
__global__ void supcon_fin_kernel(const float* __restrict__ lossAcc,
                                  const int* __restrict__ valid,
                                  float* __restrict__ out) {
  if (threadIdx.x == 0) {
    float s = 0.0f, vb = 0.0f;
    for (int i = 0; i < B_; ++i) {
      s  += lossAcc[i];
      vb += valid[i] ? 1.0f : 0.0f;
    }
    out[0] = s / fmaxf(vb, 1.0f);
  }
}

// ---------------------------------------------------------------------------
extern "C" void kernel_launch(void* const* d_in, const int* in_sizes, int n_in,
                              void* d_out, int out_size, void* d_ws, size_t ws_size,
                              hipStream_t stream) {
  (void)in_sizes; (void)n_in; (void)out_size; (void)ws_size;
  const float* feat  = (const float*)d_in[0];   // (B,T,D) f32
  const float* masks = (const float*)d_in[1];   // (B,1,T,T) f32
  const int*   icl   = (const int*)d_in[2];     // (B,) int32
  float*       out   = (float*)d_out;

  char*  ws    = (char*)d_ws;
  size_t nElem = (size_t)B_ * T_ * D_;          // 8M elements
  __bf16* hiA     = (__bf16*)ws;                             // 16 MB
  __bf16* loA     = (__bf16*)(ws + nElem * sizeof(__bf16));  // 16 MB
  float*  lossAcc = (float*)(ws + 2 * nElem * sizeof(__bf16));
  int*    valid   = (int*)(ws + 2 * nElem * sizeof(__bf16) + B_ * sizeof(float));

  supcon_init_kernel<<<1, 64, 0, stream>>>(lossAcc, valid);
  supcon_cvt_kernel<<<(unsigned)(nElem / 256), 256, 0, stream>>>(feat, hiA, loA, (long)nElem);
  supcon_valid_kernel<<<dim3(64, B_), 256, 0, stream>>>(masks, icl, valid);
  supcon_main_kernel<<<dim3(T_ / 16, B_), 256, 0, stream>>>(hiA, loA, masks, icl, lossAcc);
  supcon_fin_kernel<<<1, 32, 0, stream>>>(lossAcc, valid, out);
}
